// CustomTransformerLayer_25615184954087
// MI455X (gfx1250) — matrix operand order repaired and checked by reading, not compiled, not run on hardware
//
#include <hip/hip_runtime.h>
#include <hip/hip_bf16.h>

typedef __attribute__((ext_vector_type(16))) __bf16   v16bf;
typedef __attribute__((ext_vector_type(8)))  float    v8f;
typedef __attribute__((ext_vector_type(4)))  unsigned int v4u;

#define DEV __device__ __forceinline__

DEV unsigned short f2bf(float f) {
    unsigned int u = __builtin_bit_cast(unsigned int, f);
    unsigned int r = (u + 0x7FFFu + ((u >> 16) & 1u)) >> 16;  // RNE
    return (unsigned short)r;
}
DEV unsigned int pack2(float lo, float hi) {
    return (unsigned int)f2bf(lo) | ((unsigned int)f2bf(hi) << 16);
}

static constexpr int BM = 128, BN = 128, BK = 64;
static constexpr int LSTR = 72;   // halfwords per LDS row (144B = 9*16B: aligned, bank-spread)

union FragU { v4u q[2]; v16bf h; };

// ---- global -> register staging helpers (each thread: 32 elems of one row) ----
template<bool A_BF16>
DEV void load_stageA(v4u (&q)[4], const float* Af, const unsigned short* Ah,
                     int m0, int lda, int k0, int tid) {
    const int row = tid >> 1;
    const int col = (tid & 1) * 32;
    if constexpr (A_BF16) {
        const v4u* src = (const v4u*)(Ah + (long long)(m0 + row) * lda + k0 + col);
#pragma unroll
        for (int i = 0; i < 4; ++i) q[i] = src[i];
    } else {
        const float4* src = (const float4*)(Af + (long long)(m0 + row) * lda + k0 + col);
#pragma unroll
        for (int i = 0; i < 4; ++i) {
            float4 a = src[2 * i], b = src[2 * i + 1];
            v4u t = { pack2(a.x, a.y), pack2(a.z, a.w), pack2(b.x, b.y), pack2(b.z, b.w) };
            q[i] = t;
        }
    }
}
DEV void store_stageA(const v4u (&q)[4], unsigned short* sA, int tid) {
    const int row = tid >> 1;
    const int col = (tid & 1) * 32;
    unsigned short* dst = sA + row * LSTR + col;
#pragma unroll
    for (int i = 0; i < 4; ++i) *(v4u*)(dst + 8 * i) = q[i];
}

template<bool B_BF16, bool B_TRANS>
DEV void load_stageB(v4u (&q)[4], const float* Bf, const unsigned short* Bh,
                     int n0, int ldb, int k0, int tid) {
    const int n  = tid & 127;
    const int kb = (tid >> 7) * 32;
    if constexpr (B_TRANS) {
        if constexpr (B_BF16) {
            const v4u* src = (const v4u*)(Bh + (long long)(n0 + n) * ldb + k0 + kb);
#pragma unroll
            for (int i = 0; i < 4; ++i) q[i] = src[i];
        } else {
            const float4* src = (const float4*)(Bf + (long long)(n0 + n) * ldb + k0 + kb);
#pragma unroll
            for (int i = 0; i < 4; ++i) {
                float4 a = src[2 * i], b = src[2 * i + 1];
                v4u t = { pack2(a.x, a.y), pack2(a.z, a.w), pack2(b.x, b.y), pack2(b.z, b.w) };
                q[i] = t;
            }
        }
    } else {
        unsigned int qq[16];
#pragma unroll
        for (int i = 0; i < 16; ++i) {
            if constexpr (B_BF16) {
                unsigned short lo = Bh[(long long)(k0 + kb + 2 * i)     * ldb + n0 + n];
                unsigned short hi = Bh[(long long)(k0 + kb + 2 * i + 1) * ldb + n0 + n];
                qq[i] = (unsigned int)lo | ((unsigned int)hi << 16);
            } else {
                float lo = Bf[(long long)(k0 + kb + 2 * i)     * ldb + n0 + n];
                float hi = Bf[(long long)(k0 + kb + 2 * i + 1) * ldb + n0 + n];
                qq[i] = pack2(lo, hi);
            }
        }
#pragma unroll
        for (int i = 0; i < 4; ++i) {
            v4u t = { qq[4 * i], qq[4 * i + 1], qq[4 * i + 2], qq[4 * i + 3] };
            q[i] = t;
        }
    }
}
DEV void store_stageB(const v4u (&q)[4], unsigned short* sB, int tid) {
    const int n  = tid & 127;
    const int kb = (tid >> 7) * 32;
    unsigned short* dst = sB + n * LSTR + kb;
#pragma unroll
    for (int i = 0; i < 4; ++i) *(v4u*)(dst + 8 * i) = q[i];
}

// ---------------------------------------------------------------------------
// Tiled WMMA GEMM:  Out[M,N] = act(A[M,K] * B + bias + residual)
//   BK=64: 16 wmma per stage; ping-pong LDS double buffering (1 barrier/step)
//   8 waves per 256-thread block; wave grid 4(M) x 2(N); each wave: 2x4 tiles
// ---------------------------------------------------------------------------
template<bool A_BF16, bool B_BF16, bool B_TRANS, bool RELU, bool HAS_RES,
         bool OUT_F32, bool OUT_BF16>
__global__ __launch_bounds__(256)
void gemm_wmma_kernel(const void* __restrict__ Abase, long long aBatch, int lda,
                      const void* __restrict__ Bbase, long long bBatch, int ldb,
                      const float* __restrict__ bias,
                      const float* __restrict__ ResBase, long long resBatch,
                      float* __restrict__ OutFBase,
                      unsigned short* __restrict__ OutBBase, long long oBatch,
                      int M, int N, int K)
{
    __shared__ unsigned short sA[2][BM * LSTR];
    __shared__ unsigned short sB[2][BN * LSTR];

    const int tid   = threadIdx.x;
    const int lane  = tid & 31;
    const int wave  = tid >> 5;
    const int waveM = wave >> 1;   // 0..3
    const int waveN = wave & 1;    // 0..1
    const int m0 = blockIdx.y * BM;
    const int n0 = blockIdx.x * BN;
    const long long bz = blockIdx.z;

    const float*          Af = (const float*)Abase          + bz * aBatch;
    const unsigned short* Ah = (const unsigned short*)Abase + bz * aBatch;
    const float*          Bf = (const float*)Bbase          + bz * bBatch;
    const unsigned short* Bh = (const unsigned short*)Bbase + bz * bBatch;

    const float* Res = nullptr;
    if constexpr (HAS_RES) Res = ResBase + bz * resBatch;
    float* OutF = nullptr;
    if constexpr (OUT_F32)  OutF = OutFBase + bz * oBatch;
    unsigned short* OutB = nullptr;
    if constexpr (OUT_BF16) OutB = OutBBase + bz * oBatch;

    const v8f vzero = {0.f,0.f,0.f,0.f,0.f,0.f,0.f,0.f};
    v8f acc[2][4];
#pragma unroll
    for (int i = 0; i < 2; ++i)
#pragma unroll
        for (int j = 0; j < 4; ++j) acc[i][j] = vzero;

    // fragment base indices (CDNA5 ISA 7.12.2 layouts, 16-bit A/B)
    const int am   = waveM * 32 + (lane & 15);
    const int akb  = (lane < 16) ? 0 : 8;    // A: lanes16-31 hold K=8..15 / 24..31
    const int bnn  = waveN * 64 + (lane & 15);
    const int bkb2 = (lane < 16) ? 0 : 8;    // B: lanes16-31 hold K=8..15 / 24..31

    // ---- prologue: stage tile 0 into buffer 0 ----
    v4u aq[4], bq[4];
    load_stageA<A_BF16>(aq, Af, Ah, m0, lda, 0, tid);
    load_stageB<B_BF16, B_TRANS>(bq, Bf, Bh, n0, ldb, 0, tid);
    store_stageA(aq, sA[0], tid);
    store_stageB(bq, sB[0], tid);
    __syncthreads();

    int p = 0;
    for (int k0 = 0; k0 < K; k0 += BK) {
        const unsigned short* cA = sA[p];
        const unsigned short* cB = sB[p];
        const int knext = k0 + BK;

        v16bf aF[2], bF[4];

        // ---- K half 0: fragments ----
#pragma unroll
        for (int mi = 0; mi < 2; ++mi) {
            FragU u;
            u.q[0] = *(const v4u*)(cA + (am + mi * 16) * LSTR + akb);
            u.q[1] = *(const v4u*)(cA + (am + mi * 16) * LSTR + akb + 16);
            aF[mi] = u.h;
        }
#pragma unroll
        for (int ni = 0; ni < 4; ++ni) {
            FragU u;
            u.q[0] = *(const v4u*)(cB + (bnn + ni * 16) * LSTR + bkb2);
            u.q[1] = *(const v4u*)(cB + (bnn + ni * 16) * LSTR + bkb2 + 16);
            bF[ni] = u.h;
        }

        // ---- issue next tile's global loads (hidden behind the WMMAs) ----
        if (knext < K) {
            load_stageA<A_BF16>(aq, Af, Ah, m0, lda, knext, tid);
            load_stageB<B_BF16, B_TRANS>(bq, Bf, Bh, n0, ldb, knext, tid);
            if (knext + BK < K) {   // prefetch tile after next -> global_prefetch_b8
                if constexpr (A_BF16)
                    __builtin_prefetch(Ah + (long long)(m0 + (tid >> 1)) * lda + knext + BK, 0, 1);
                else
                    __builtin_prefetch(Af + (long long)(m0 + (tid >> 1)) * lda + knext + BK, 0, 1);
            }
        }

#pragma unroll
        for (int mi = 0; mi < 2; ++mi)
#pragma unroll
            for (int ni = 0; ni < 4; ++ni)
                acc[mi][ni] = __builtin_amdgcn_wmma_f32_16x16x32_bf16(
                    false, aF[mi], false, bF[ni], (short)0, acc[mi][ni], false, false);

        // ---- K half 1: fragments (offset +32 in K) ----
#pragma unroll
        for (int mi = 0; mi < 2; ++mi) {
            FragU u;
            u.q[0] = *(const v4u*)(cA + (am + mi * 16) * LSTR + 32 + akb);
            u.q[1] = *(const v4u*)(cA + (am + mi * 16) * LSTR + 32 + akb + 16);
            aF[mi] = u.h;
        }
#pragma unroll
        for (int ni = 0; ni < 4; ++ni) {
            FragU u;
            u.q[0] = *(const v4u*)(cB + (bnn + ni * 16) * LSTR + 32 + bkb2);
            u.q[1] = *(const v4u*)(cB + (bnn + ni * 16) * LSTR + 32 + bkb2 + 16);
            bF[ni] = u.h;
        }

#pragma unroll
        for (int mi = 0; mi < 2; ++mi)
#pragma unroll
            for (int ni = 0; ni < 4; ++ni)
                acc[mi][ni] = __builtin_amdgcn_wmma_f32_16x16x32_bf16(
                    false, aF[mi], false, bF[ni], (short)0, acc[mi][ni], false, false);

        // ---- commit next tile into the other buffer, single barrier ----
        if (knext < K) {
            store_stageA(aq, sA[p ^ 1], tid);
            store_stageB(bq, sB[p ^ 1], tid);
        }
        __syncthreads();
        p ^= 1;
    }

    // ---- epilogue: C/D layout = 8 VGPRs, lanes16-31 -> M+8 ----
#pragma unroll
    for (int mi = 0; mi < 2; ++mi) {
#pragma unroll
        for (int ni = 0; ni < 4; ++ni) {
            const int rowBase = m0 + waveM * 32 + mi * 16 + ((lane < 16) ? 0 : 8);
            const int ncol    = n0 + waveN * 64 + ni * 16 + (lane & 15);
            const float bv = bias ? bias[ncol] : 0.0f;
#pragma unroll
            for (int r = 0; r < 8; ++r) {
                const long long idx = (long long)(rowBase + r) * N + ncol;
                float v = acc[mi][ni][r] + bv;
                if constexpr (HAS_RES) v += Res[idx];
                if constexpr (RELU)    v = fmaxf(v, 0.0f);
                if constexpr (OUT_F32)  OutF[idx] = v;
                if constexpr (OUT_BF16) OutB[idx] = f2bf(v);
            }
        }
    }
}

// ---------------------------------------------------------------------------
// Row softmax: one 256-thread block per row of 2048, writes bf16 probs
// ---------------------------------------------------------------------------
__global__ __launch_bounds__(256)
void softmax_kernel(const float* __restrict__ S, unsigned short* __restrict__ P, int cols)
{
    __shared__ float red[256];
    const long long row = blockIdx.x;
    const int tid = threadIdx.x;
    const float* s = S + row * cols;

    float v[8];
    float lmax = -3.4e38f;
#pragma unroll
    for (int i = 0; i < 8; ++i) { v[i] = s[tid + (i << 8)]; lmax = fmaxf(lmax, v[i]); }
    red[tid] = lmax; __syncthreads();
    for (int st = 128; st > 0; st >>= 1) {
        if (tid < st) red[tid] = fmaxf(red[tid], red[tid + st]);
        __syncthreads();
    }
    const float mx = red[0]; __syncthreads();

    float lsum = 0.f;
#pragma unroll
    for (int i = 0; i < 8; ++i) { v[i] = __expf(v[i] - mx); lsum += v[i]; }
    red[tid] = lsum; __syncthreads();
    for (int st = 128; st > 0; st >>= 1) {
        if (tid < st) red[tid] += red[tid + st];
        __syncthreads();
    }
    const float inv = 1.0f / red[0];
#pragma unroll
    for (int i = 0; i < 8; ++i) P[row * cols + tid + (i << 8)] = f2bf(v[i] * inv);
}

// ---------------------------------------------------------------------------
// LayerNorm over rows of 1024; writes fp32 (and optional bf16 copy)
// ---------------------------------------------------------------------------
__global__ __launch_bounds__(256)
void layernorm_kernel(const float* __restrict__ X, const float* __restrict__ g,
                      const float* __restrict__ be, float* __restrict__ OutF,
                      unsigned short* __restrict__ OutB, int cols)
{
    __shared__ float red[256];
    const long long row = blockIdx.x;
    const int tid = threadIdx.x;
    const float* x = X + row * cols;

    float v[4];
    float s = 0.f;
#pragma unroll
    for (int i = 0; i < 4; ++i) { v[i] = x[tid + (i << 8)]; s += v[i]; }
    red[tid] = s; __syncthreads();
    for (int st = 128; st > 0; st >>= 1) {
        if (tid < st) red[tid] += red[tid + st];
        __syncthreads();
    }
    const float mean = red[0] / (float)cols; __syncthreads();

    float vs = 0.f;
#pragma unroll
    for (int i = 0; i < 4; ++i) { const float d = v[i] - mean; vs += d * d; }
    red[tid] = vs; __syncthreads();
    for (int st = 128; st > 0; st >>= 1) {
        if (tid < st) red[tid] += red[tid + st];
        __syncthreads();
    }
    const float rstd = rsqrtf(red[0] / (float)cols + 1e-5f);

#pragma unroll
    for (int i = 0; i < 4; ++i) {
        const int c = tid + (i << 8);
        const float y = (v[i] - mean) * rstd * g[c] + be[c];
        OutF[row * cols + c] = y;
        if (OutB) OutB[row * cols + c] = f2bf(y);
    }
}

// ---------------------------------------------------------------------------
extern "C" void kernel_launch(void* const* d_in, const int* in_sizes, int n_in,
                              void* d_out, int out_size, void* d_ws, size_t ws_size,
                              hipStream_t stream) {
    (void)in_sizes; (void)n_in; (void)out_size; (void)ws_size;

    const int   Bc = 4, S = 2048, E = 1024, FFD = 4096;
    const long long BS = (long long)Bc * S;   // 8192

    const float* src = (const float*)d_in[0];
    const float* Wq  = (const float*)d_in[1];  const float* bq = (const float*)d_in[2];
    const float* Wk  = (const float*)d_in[3];  const float* bk = (const float*)d_in[4];
    const float* Wv  = (const float*)d_in[5];  const float* bv = (const float*)d_in[6];
    const float* Wo  = (const float*)d_in[7];  const float* bo = (const float*)d_in[8];
    const float* W1  = (const float*)d_in[9];  const float* b1 = (const float*)d_in[10];
    const float* W2  = (const float*)d_in[11]; const float* b2 = (const float*)d_in[12];
    const float* g1  = (const float*)d_in[13]; const float* be1 = (const float*)d_in[14];
    const float* g2  = (const float*)d_in[15]; const float* be2 = (const float*)d_in[16];

    // workspace carve-up (256B aligned)
    char* w = (char*)d_ws;
    auto carve = [&](size_t bytes) { void* p = (void*)w; w += (bytes + 255) & ~(size_t)255; return p; };
    unsigned short* Qb    = (unsigned short*)carve(BS * E * 2);
    unsigned short* Kb    = (unsigned short*)carve(BS * E * 2);
    unsigned short* Vb    = (unsigned short*)carve(BS * E * 2);
    float*          Sc    = (float*)carve((long long)Bc * S * S * 4);
    unsigned short* Pb    = (unsigned short*)carve((long long)Bc * S * S * 2);
    unsigned short* attnB = (unsigned short*)carve(BS * E * 2);
    float*          y1    = (float*)carve(BS * E * 4);
    float*          x     = (float*)carve(BS * E * 4);
    unsigned short* xb    = (unsigned short*)carve(BS * E * 2);
    unsigned short* h     = (unsigned short*)carve(BS * FFD * 2);
    float*          y2    = (float*)carve(BS * E * 4);

    const dim3 blk(256);

    // 1) Q, K, V projections: [8192,1024] = src * W + b  -> bf16
    {
        dim3 grd(E / BN, BS / BM, 1);
        gemm_wmma_kernel<false,false,false,false,false,false,true><<<grd, blk, 0, stream>>>(
            src, 0, E, Wq, 0, E, bq, nullptr, 0, nullptr, Qb, 0, (int)BS, E, E);
        gemm_wmma_kernel<false,false,false,false,false,false,true><<<grd, blk, 0, stream>>>(
            src, 0, E, Wk, 0, E, bk, nullptr, 0, nullptr, Kb, 0, (int)BS, E, E);
        gemm_wmma_kernel<false,false,false,false,false,false,true><<<grd, blk, 0, stream>>>(
            src, 0, E, Wv, 0, E, bv, nullptr, 0, nullptr, Vb, 0, (int)BS, E, E);
    }

    // 2) scores[b] = Q[b] * K[b]^T  (K pre-transposed layout [N=S][K=E]) -> fp32
    {
        dim3 grd(S / BN, S / BM, Bc);
        gemm_wmma_kernel<true,true,true,false,false,true,false><<<grd, blk, 0, stream>>>(
            Qb, (long long)S * E, E, Kb, (long long)S * E, E, nullptr,
            nullptr, 0, Sc, nullptr, (long long)S * S, S, S, E);
    }

    // 3) softmax rows -> bf16 P
    softmax_kernel<<<dim3((unsigned)BS), blk, 0, stream>>>(Sc, Pb, S);

    // 4) attn[b] = P[b] * V[b]  -> bf16
    {
        dim3 grd(E / BN, S / BM, Bc);
        gemm_wmma_kernel<true,true,false,false,false,false,true><<<grd, blk, 0, stream>>>(
            Pb, (long long)S * S, S, Vb, (long long)S * E, E, nullptr,
            nullptr, 0, nullptr, attnB, (long long)S * E, S, E, S);
    }

    // 5) y1 = attn * Wo + bo + src  -> fp32
    {
        dim3 grd(E / BN, BS / BM, 1);
        gemm_wmma_kernel<true,false,false,false,true,true,false><<<grd, blk, 0, stream>>>(
            attnB, 0, E, Wo, 0, E, bo, src, 0, y1, nullptr, 0, (int)BS, E, E);
    }

    // 6) x = LN(y1)*g1+be1  -> fp32 + bf16
    layernorm_kernel<<<dim3((unsigned)BS), blk, 0, stream>>>(y1, g1, be1, x, xb, E);

    // 7) h = relu(x * W1 + b1)  -> bf16
    {
        dim3 grd(FFD / BN, BS / BM, 1);
        gemm_wmma_kernel<true,false,false,true,false,false,true><<<grd, blk, 0, stream>>>(
            xb, 0, E, W1, 0, FFD, b1, nullptr, 0, nullptr, h, 0, (int)BS, FFD, E);
    }

    // 8) y2 = h * W2 + b2 + x  -> fp32
    {
        dim3 grd(E / BN, BS / BM, 1);
        gemm_wmma_kernel<true,false,false,false,true,true,false><<<grd, blk, 0, stream>>>(
            h, 0, FFD, W2, 0, E, b2, x, 0, y2, nullptr, 0, (int)BS, E, FFD);
    }

    // 9) out = LN(y2)*g2+be2  -> fp32 d_out
    layernorm_kernel<<<dim3((unsigned)BS), blk, 0, stream>>>(y2, g2, be2, (float*)d_out, nullptr, E);
}
